// LSTM_29446295781824
// MI455X (gfx1250) — compile-verified, matrix operand order checked by reading, not run on hardware
//
#include <hip/hip_runtime.h>

typedef __bf16 bf16_t;
typedef __attribute__((ext_vector_type(16))) __bf16 v16bf;
typedef __attribute__((ext_vector_type(8)))  __bf16 bf16x8;
typedef __attribute__((ext_vector_type(8)))  float  v8f;
typedef __attribute__((ext_vector_type(4)))  float  f32x4;

union ABfrag { v16bf v; bf16x8 h[2]; };

__device__ __forceinline__ bf16_t f2bf(float f) {
  union { float f; unsigned u; } x; x.f = f;
  unsigned r = x.u + 0x7FFFu + ((x.u >> 16) & 1u);   // round-nearest-even
  unsigned short hs = (unsigned short)(r >> 16);
  bf16_t o; __builtin_memcpy(&o, &hs, 2); return o;
}
__device__ __forceinline__ float sigm(float x)      { return 1.0f / (1.0f + __expf(-x)); }
__device__ __forceinline__ float tanh_fast(float x) { return 2.0f / (1.0f + __expf(-2.0f * x)) - 1.0f; }

constexpr int BATCH = 256, T = 2048, H = 128, G = 512;

// Persistent per-workgroup LSTM layer.
//   Each block owns 16 batch rows for all 2048 timesteps.
//   A = [input_t | h] (16 x KTOT, bf16 in LDS), B = [W_ih^T ; W_hh^T] pre-swizzled
//   into WMMA B-fragment order in LDS. gates (16x512 f32) in LDS, c-state in regs.
template<int KCH, int KINPAD, int INFEAT, bool LOAD_SEQ, bool STORE_SEQ>
__global__ __launch_bounds__(256) void lstm_layer(
    const float*  __restrict__ xin,     // layer0 raw input (f32) or nullptr
    const bf16_t* __restrict__ seq_in,  // layer1 input sequence (bf16) or nullptr
    const float*  __restrict__ W_ih, const float* __restrict__ W_hh,
    const float*  __restrict__ b_ih, const float* __restrict__ b_hh,
    bf16_t* __restrict__ seq_out,       // layer0: full h sequence (bf16)
    float*  __restrict__ h_final)       // layer1: final h (f32)
{
  constexpr int KTOT = KCH * 32;
  extern __shared__ __align__(16) char smem[];
  bf16_t* Wfrag  = (bf16_t*)smem;                                              // 32*KCH frags * 512 elem
  float*  gatesL = (float*)(smem + (size_t)KCH * 32768);                       // 16 x 512 f32
  bf16_t* Astage = (bf16_t*)(smem + (size_t)KCH * 32768 + 32768);              // 16 x KTOT bf16
  float*  biasL  = (float*)(smem + (size_t)KCH * 32768 + 32768 + (size_t)KCH * 1024); // 512 f32

  const int tid    = threadIdx.x;
  const int wgbase = blockIdx.x * 16;

  // ---- one-time setup: combined bias, zeroed A staging (h0=0 + zero padding)
  for (int n = tid; n < G; n += 256) biasL[n] = b_ih[n] + b_hh[n];
  for (int i = tid; i < 16 * KTOT; i += 256) Astage[i] = f2bf(0.0f);

  // ---- swizzle [W_ih^T ; W_hh^T] (f32 -> bf16) into WMMA B-fragment lane order
  const int E = 32 * KCH * 512;
  for (int e = tid; e < E; e += 256) {
    int frag = e >> 9, rem = e & 511, lane = rem >> 4, idx = rem & 15;
    int ntile = frag / KCH, kch = frag % KCH;
    int koff = (lane < 16) ? ((idx < 8) ? idx : idx + 8)
                           : ((idx < 8) ? idx + 8 : idx + 16);
    int k = kch * 32 + koff;
    int n = ntile * 16 + (lane & 15);
    float v;
    if (k < KINPAD) v = (k < INFEAT) ? W_ih[n * INFEAT + k] : 0.0f;
    else            v = W_hh[n * H + (k - KINPAD)];
    Wfrag[e] = f2bf(v);
  }

  const int lane = tid & 31, wave = tid >> 5;
  const int mrow = lane & 15, hi = lane >> 4;       // WMMA A/C lane decomposition
  const int pm = tid >> 4, pj = (tid & 15) * 8;     // pointwise: row, col-base
  float cst[8], hreg[8];
  #pragma unroll
  for (int q = 0; q < 8; ++q) { cst[q] = 0.0f; hreg[q] = 0.0f; }

  __syncthreads();

  // ---- hoist loop-invariant bias (one value per lane per owned N-tile)
  float bsv[4];
  #pragma unroll
  for (int nt = 0; nt < 4; ++nt)
    bsv[nt] = biasL[(wave * 4 + nt) * 16 + (lane & 15)];

  for (int t = 0; t < T; ++t) {
    // ---- stage the input part of A for this timestep
    if constexpr (LOAD_SEQ) {
      int r = tid >> 4, off = (tid & 15) * 8;
      const bf16x8* src = (const bf16x8*)(seq_in + ((size_t)(wgbase + r) * T + t) * H + off);
      *(bf16x8*)&Astage[r * KTOT + off] = *src;
      if (t + 1 < T) __builtin_prefetch((const char*)src + H * 2, 0, 0); // next step
    } else {
      if (tid < 16 * INFEAT) {
        int r = tid / INFEAT, i = tid % INFEAT;
        Astage[r * KTOT + i] = f2bf(xin[((size_t)(wgbase + r) * T + t) * INFEAT + i]);
      }
    }
    __syncthreads();

    // ---- A fragments: shared across all 4 N-tiles of this wave, load once
    ABfrag afr[KCH];
    #pragma unroll
    for (int kc = 0; kc < KCH; ++kc) {
      const bf16_t* ap = &Astage[mrow * KTOT + kc * 32 + hi * 8];
      afr[kc].h[0] = *(const bf16x8*)ap;            // K {0..7} / {8..15}
      afr[kc].h[1] = *(const bf16x8*)(ap + 16);     // K {16..23} / {24..31}
    }

    // ---- gates = [x_t | h] @ [W_ih^T ; W_hh^T] + bias   via v_wmma bf16
    #pragma unroll
    for (int nt = 0; nt < 4; ++nt) {
      const int ntile = wave * 4 + nt;
      const int ncol  = ntile * 16 + (lane & 15);
      // issue all B-fragment loads up front so they overlap the wmma chain
      ABfrag bfr[KCH];
      #pragma unroll
      for (int kc = 0; kc < KCH; ++kc) {
        const bf16_t* bp = &Wfrag[((size_t)(ntile * KCH + kc)) * 512 + lane * 16];
        bfr[kc].h[0] = *(const bf16x8*)bp;
        bfr[kc].h[1] = *(const bf16x8*)(bp + 8);
      }
      const float bs = bsv[nt];
      v8f acc = {bs, bs, bs, bs, bs, bs, bs, bs};
      #pragma unroll
      for (int kc = 0; kc < KCH; ++kc)
        acc = __builtin_amdgcn_wmma_f32_16x16x32_bf16(false, afr[kc].v, false, bfr[kc].v,
                                                      (short)0, acc, false, false);
      #pragma unroll
      for (int v = 0; v < 8; ++v)
        gatesL[(hi * 8 + v) * G + ncol] = acc[v];
    }
    __syncthreads();

    // ---- pointwise LSTM cell (vectorized b128 gate reads)
    const f32x4* gi = (const f32x4*)&gatesL[pm * G + pj];
    f32x4 ivv[2], fvv[2], gvv[2], ovv[2];
    #pragma unroll
    for (int p = 0; p < 2; ++p) {
      ivv[p] = gi[p];                 // i  : cols [0,128)
      fvv[p] = gi[p + H / 4];         // f  : cols [128,256)
      gvv[p] = gi[p + 2 * H / 4];     // g  : cols [256,384)
      ovv[p] = gi[p + 3 * H / 4];     // o  : cols [384,512)
    }
    #pragma unroll
    for (int q = 0; q < 8; ++q) {
      float iv = sigm(ivv[q >> 2][q & 3]);
      float fv = sigm(fvv[q >> 2][q & 3]);
      float gv = tanh_fast(gvv[q >> 2][q & 3]);
      float ov = sigm(ovv[q >> 2][q & 3]);
      float cn = fv * cst[q] + iv * gv;
      cst[q] = cn;
      float hn = ov * tanh_fast(cn);
      hreg[q] = hn;
    }
    // write new h back into A staging (bf16), one b128 store
    {
      bf16x8 hv;
      #pragma unroll
      for (int q = 0; q < 8; ++q) hv[q] = f2bf(hreg[q]);
      *(bf16x8*)&Astage[pm * KTOT + KINPAD + pj] = hv;
      if constexpr (STORE_SEQ)
        *(bf16x8*)(seq_out + ((size_t)(wgbase + pm) * T + t) * H + pj) = hv;
    }
  }

  if (h_final) {
    #pragma unroll
    for (int q = 0; q < 8; ++q)
      h_final[(size_t)(wgbase + pm) * H + pj + q] = hreg[q];
  }
}

// dropout (deterministic hash mask, p=0.3, scale 1/0.7) + 128->7 linear head
__global__ __launch_bounds__(256) void head_kernel(
    const float* __restrict__ h2, const float* __restrict__ Wout,
    const float* __restrict__ bout, float* __restrict__ out)
{
  int cls = blockIdx.x;   // 0..6
  int b   = threadIdx.x;  // 0..255
  const float scale = 1.0f / 0.7f;
  float s = 0.0f;
  for (int j = 0; j < H; ++j) {
    unsigned idx = (unsigned)(b * H + j);
    unsigned u = idx * 2654435761u; u ^= u >> 16; u *= 2246822519u; u ^= u >> 13;
    float keep = ((u & 0xFFFFFFu) < 11744051u) ? scale : 0.0f; // 0.7 * 2^24
    s = fmaf(keep * h2[b * H + j], Wout[cls * H + j], s);
  }
  out[b * 7 + cls] = s + bout[cls];
}

extern "C" void kernel_launch(void* const* d_in, const int* in_sizes, int n_in,
                              void* d_out, int out_size, void* d_ws, size_t ws_size,
                              hipStream_t stream) {
  (void)in_sizes; (void)n_in; (void)out_size; (void)ws_size;
  const float* x    = (const float*)d_in[0];
  const float* Wih0 = (const float*)d_in[1];
  const float* Whh0 = (const float*)d_in[2];
  const float* bih0 = (const float*)d_in[3];
  const float* bhh0 = (const float*)d_in[4];
  const float* Wih1 = (const float*)d_in[5];
  const float* Whh1 = (const float*)d_in[6];
  const float* bih1 = (const float*)d_in[7];
  const float* bhh1 = (const float*)d_in[8];
  const float* Wout = (const float*)d_in[9];
  const float* bout = (const float*)d_in[10];

  bf16_t* h1seq = (bf16_t*)d_ws;                                     // 256*2048*128 bf16 = 128 MB
  float*  h2fin = (float*)((char*)d_ws + (size_t)BATCH * T * H * 2); // 256*128 f32

  size_t smem0 = (size_t)5 * 32768 + 32768 + (size_t)5 * 1024 + 2048; // ~200 KB
  size_t smem1 = (size_t)8 * 32768 + 32768 + (size_t)8 * 1024 + 2048; // ~305 KB (fits 320 KB WGP LDS)

  // layer 0: K = 32(pad of 5 feats) + 128(h) = 160
  lstm_layer<5, 32, 5, false, true><<<16, 256, smem0, stream>>>(
      x, nullptr, Wih0, Whh0, bih0, bhh0, h1seq, nullptr);
  // layer 1: K = 128(input h1) + 128(h) = 256, fused input+recurrent GEMM
  lstm_layer<8, 128, 128, true, false><<<16, 256, smem1, stream>>>(
      nullptr, h1seq, Wih1, Whh1, bih1, bhh1, nullptr, h2fin);
  head_kernel<<<7, 256, 0, stream>>>(h2fin, Wout, bout, (float*)d_out);
}
